// HandwritingRNN_35519379538494
// MI455X (gfx1250) — matrix operand order repaired and checked by reading, not compiled, not run on hardware
//
#include <hip/hip_runtime.h>
#include <stdint.h>
#include <stddef.h>

// Problem dims
#define Bn    128
#define Tn    800
#define Un    64
#define VOCABn 80
#define Hn    400
#define Kmix  10
#define OUTn  121

// Padded activation K layout: [ x(3) | w(80) | pad(13) | h(400) | pad(16) ] = 512
#define KDIM  512
#define XWK   96          // x+w section (k < 96), h starts at k=96
#define HPAD  416         // padded h row stride (bf16)
#define NWG   25          // 25 persistent workgroups, 16 h-columns each
#define HCOLS 16

typedef __attribute__((ext_vector_type(16))) __bf16 bf16x16;
typedef __attribute__((ext_vector_type(8)))  float  f32x8;
typedef __attribute__((ext_vector_type(4)))  unsigned int v4u;
typedef __attribute__((ext_vector_type(8)))  int v8i;
typedef __attribute__((ext_vector_type(4)))  int v4i;

union FragA { uint4 u[2]; bf16x16 v; };

__device__ __forceinline__ unsigned short f2bf(float f) {
  unsigned u = __float_as_uint(f);
  u += 0x7FFFu + ((u >> 16) & 1u);        // round-to-nearest-even
  return (unsigned short)(u >> 16);
}
__device__ __forceinline__ float sigmoidf(float x) { return 1.0f / (1.0f + __expf(-x)); }

__device__ __forceinline__ void grid_barrier(unsigned* ctr, unsigned target) {
  __syncthreads();
  if (threadIdx.x == 0) {
    __threadfence();
    __hip_atomic_fetch_add(ctr, 1u, __ATOMIC_ACQ_REL, __HIP_MEMORY_SCOPE_AGENT);
    while (__hip_atomic_load(ctr, __ATOMIC_ACQUIRE, __HIP_MEMORY_SCOPE_AGENT) < target)
      __builtin_amdgcn_s_sleep(1);
  }
  __syncthreads();
}

// ---------------------------------------------------------------------------
// TDM: issue a 2-D bf16 tile copy global -> LDS via the Tensor Data Mover.
// Descriptor packing per CDNA5 ISA:
//   group0: count=1 | lds_addr | global_addr(57b) | type=2
//   group1: data_size=1(2B) | tensor_dim0/1 | tile_dim0/1 | dim0_stride
//   groups 2/3 (+spare group): zero (tensor <= 2D)
// LDS offset = low 32 bits of the generic pointer (LDS aperture maps addr[31:0]).
// All operands are wave-uniform -> SGPR groups. Tracked with TENSORcnt.
// This toolchain declares the 6-arg builtin:
//   (v4u g0, v8i g1, v4i g2, v4i g3, v8i extra, i32 cpol)
// ---------------------------------------------------------------------------
__device__ __forceinline__ void tdm_load_tile_2d(void* lds_dst, const void* gsrc,
                                                 unsigned tile0, unsigned tile1,
                                                 unsigned stride0_elems) {
  unsigned long long ga = (unsigned long long)(uintptr_t)gsrc;
  unsigned lds_off = (unsigned)(uintptr_t)lds_dst;
  v4u g0;
  g0[0] = 1u;                                               // count=1, user mode
  g0[1] = lds_off;                                          // lds_addr (bytes)
  g0[2] = (unsigned)(ga & 0xFFFFFFFFu);                     // global_addr[31:0]
  g0[3] = (unsigned)((ga >> 32) & 0x01FFFFFFu) | (2u << 30);// addr[56:32] | type=2
  v8i g1;
  g1[0] = (int)(1u << 16);                                  // data_size=1 -> 2 bytes
  g1[1] = (int)((tile0 & 0xFFFFu) << 16);                   // tensor_dim0[15:0]
  g1[2] = (int)(((tile0 >> 16) & 0xFFFFu) |                 // tensor_dim0[31:16]
                ((tile1 & 0xFFFFu) << 16));                 // tensor_dim1[15:0]
  g1[3] = (int)(((tile1 >> 16) & 0xFFFFu) |                 // tensor_dim1[31:16]
                ((tile0 & 0xFFFFu) << 16));                 // tile_dim0
  g1[4] = (int)(tile1 & 0xFFFFu);                           // tile_dim1 (tile_dim2=0)
  g1[5] = (int)stride0_elems;                               // dim0_stride[31:0]
  g1[6] = 0;                                                // stride hi / dim1_stride lo
  g1[7] = 0;
  v4i z4; z4[0] = 0; z4[1] = 0; z4[2] = 0; z4[3] = 0;
  v8i z8;
  #pragma unroll
  for (int i = 0; i < 8; ++i) z8[i] = 0;
  __builtin_amdgcn_tensor_load_to_lds(g0, g1, z4, z4, z8, 0);
}

// ---------------------------------------------------------------------------
// Kernel 0: convert fp32 weights to bf16, n-major, zero-padded to WMMA tiles.
//   wz  : [1600][512]  (concat W_x rows 0..82 at k<83, W_h rows at k in [96,496))
//   watt: [32][512]    (W_att, n<30 valid)
//   wout: [128][416]   (W_out, n<121, k<400 valid)
// ---------------------------------------------------------------------------
__global__ void prep_weights(const float* __restrict__ Wx, const float* __restrict__ Wh,
                             const float* __restrict__ Wa, const float* __restrict__ Wo,
                             unsigned short* __restrict__ wz, unsigned short* __restrict__ watt,
                             unsigned short* __restrict__ wout) {
  const int NZ = 1600 * KDIM, NA = 32 * KDIM, NO = 128 * HPAD;
  for (int i = blockIdx.x * blockDim.x + threadIdx.x; i < NZ + NA + NO;
       i += gridDim.x * blockDim.x) {
    if (i < NZ) {
      int n = i / KDIM, k = i - n * KDIM;
      float v = 0.f;
      if (k < 83) v = Wx[k * 1600 + n];
      else if (k >= XWK && k < XWK + Hn) v = Wh[(k - XWK) * 1600 + n];
      wz[i] = f2bf(v);
    } else if (i < NZ + NA) {
      int j = i - NZ, n = j / KDIM, k = j - n * KDIM;
      float v = 0.f;
      if (n < 3 * Kmix) {
        if (k < 83) v = Wa[k * (3 * Kmix) + n];
        else if (k >= XWK && k < XWK + Hn) v = Wa[(83 + k - XWK) * (3 * Kmix) + n];
      }
      watt[j] = f2bf(v);
    } else {
      int j = i - NZ - NA, n = j / HPAD, k = j - n * HPAD;
      float v = (n < OUTn && k < Hn) ? Wo[k * OUTn + n] : 0.f;
      wout[j] = f2bf(v);
    }
  }
}

// ---------------------------------------------------------------------------
// Kernel 1: persistent recurrent kernel. 25 WGs x 256 threads (8 waves).
// Each WG owns h columns [wg*16, wg*16+16) and the 64 z columns of its 4 gates.
// cst lives in WMMA-accumulator layout registers; weights live in LDS (bf16),
// staged once by the Tensor Data Mover (4 x 16x512 bf16 tiles, one per gate).
// ---------------------------------------------------------------------------
__global__ __launch_bounds__(256) void rnn_persistent(
    const float* __restrict__ x, const int* __restrict__ c, const int* __restrict__ clen,
    const float* __restrict__ b_lstm, const float* __restrict__ b_att,
    const unsigned short* __restrict__ wz_g, const unsigned short* __restrict__ watt_g,
    unsigned short* __restrict__ hbuf, unsigned short* __restrict__ wbuf,
    unsigned short* __restrict__ hs, unsigned* __restrict__ ctr)
{
  __shared__ unsigned short sWz[64 * KDIM];    // 64 KB: this WG's weight slice
  __shared__ unsigned short sXW[Bn * XWK];     // 24 KB: [x_t | w_{t-1}] bf16
  __shared__ float sAtt[Bn * 32];              // 16 KB: attention logits (WG0)
  __shared__ float sW[Bn * VOCABn];            // 40 KB: window vector accum (WG0)
  __shared__ float sKappa[Bn * Kmix];          //  5 KB: kappa state (WG0)

  const int tid = threadIdx.x;
  const int wg  = blockIdx.x;
  const int hbase = wg * HCOLS;
  const int lane = tid & 31;
  const int wave = tid >> 5;

  // TDM-stage weight slice: LDS rows (g*16+j) <-> global rows n = g*400 + hbase + j
  // (16 contiguous n-major rows per gate -> one 2-D tile per gate).
  if (wave == 0) {
    #pragma unroll
    for (int g = 0; g < 4; ++g)
      tdm_load_tile_2d(&sWz[g * 16 * KDIM],
                       wz_g + (size_t)(g * Hn + hbase) * KDIM,
                       KDIM, 16, KDIM);
    __builtin_amdgcn_s_wait_tensorcnt(0);
  }
  if (wg == 0) for (int i = tid; i < Bn * Kmix; i += 256) sKappa[i] = 0.f;
  __syncthreads();

  const int arow = wave * HCOLS + (lane & 15);  // A operand: batch row (M = lane&15)
  const int acol = (lane >> 4) << 3;            // A chunk offset within k-tile (0|8)
  const int bk0  = (lane >> 4) << 3;            // B chunk offset within k-tile (0|8)
  const int ncol = hbase + (lane & 15);
  const float bi = b_lstm[0 * Hn + ncol];
  const float bf = b_lstm[1 * Hn + ncol];
  const float bg = b_lstm[2 * Hn + ncol];
  const float bo = b_lstm[3 * Hn + ncol];

  f32x8 cst;
  #pragma unroll
  for (int r = 0; r < 8; ++r) cst[r] = 0.f;

  for (int t = 0; t < Tn; ++t) {
    const unsigned short* hprev = hbuf + (size_t)(t & 1) * (Bn * HPAD);
    unsigned short*       hnext = hbuf + (size_t)((t + 1) & 1) * (Bn * HPAD);

    // Build [x_t | w_{t-1}] rows in LDS (w_buf is already bf16)
    for (int i = tid; i < Bn * XWK; i += 256) {
      int b = i / XWK, k = i - b * XWK;
      unsigned short v;
      if (k < 3)       v = f2bf(x[((size_t)b * Tn + t) * 3 + k]);
      else if (k < 83) v = wbuf[b * VOCABn + (k - 3)];
      else             v = 0;
      sXW[i] = v;
    }
    __syncthreads();

    // z-slice = Act(128x512) @ Wz-slice(512x64) : 4 gate accumulators / wave
    f32x8 acc[4];
    #pragma unroll
    for (int g = 0; g < 4; ++g)
      #pragma unroll
      for (int r = 0; r < 8; ++r) acc[g][r] = 0.f;

    #pragma unroll
    for (int kt = 0; kt < 16; ++kt) {
      FragA fa;
      if (kt < 3) {
        const unsigned short* p = &sXW[arow * XWK + kt * 32 + acol];
        fa.u[0] = *(const uint4*)p;
        fa.u[1] = *(const uint4*)(p + 16);
      } else {
        const unsigned short* p = hprev + (size_t)arow * HPAD + (kt * 32 - XWK) + acol;
        fa.u[0] = *(const uint4*)p;
        fa.u[1] = *(const uint4*)(p + 16);
      }
      #pragma unroll
      for (int g = 0; g < 4; ++g) {
        FragA fb;
        const unsigned short* q = &sWz[(g * 16 + (lane & 15)) * KDIM + kt * 32 + bk0];
        fb.u[0] = *(const uint4*)q;
        fb.u[1] = *(const uint4*)(q + 16);
        acc[g] = __builtin_amdgcn_wmma_f32_16x16x32_bf16(false, fa.v, false, fb.v,
                                                         (short)0, acc[g], false, false);
      }
    }

    // LSTM cell, all in accumulator layout (M = r + 8*(lane>>4), N = lane&15)
    const int mrow = wave * HCOLS + ((lane >> 4) << 3);
    #pragma unroll
    for (int r = 0; r < 8; ++r) {
      float ig = sigmoidf(acc[0][r] + bi);
      float fg = sigmoidf(acc[1][r] + bf);
      float gg = tanhf(acc[2][r] + bg);
      float og = sigmoidf(acc[3][r] + bo);
      float cs = fg * cst[r] + ig * gg;
      cst[r] = cs;
      unsigned short hb = f2bf(og * tanhf(cs));
      int row = mrow + r;
      hnext[(size_t)row * HPAD + ncol] = hb;
      hs[((size_t)row * Tn + t) * Hn + ncol] = hb;
    }

    grid_barrier(ctr, (unsigned)(NWG * (2 * t + 1)));   // h_t published

    if (wg == 0) {
      // att = [x_t, w_{t-1}, h_t] @ W_att  (N padded 30->32), WMMA again
      f32x8 aat[2];
      #pragma unroll
      for (int nt = 0; nt < 2; ++nt)
        #pragma unroll
        for (int r = 0; r < 8; ++r) aat[nt][r] = 0.f;

      #pragma unroll
      for (int kt = 0; kt < 16; ++kt) {
        FragA fa;
        if (kt < 3) {
          const unsigned short* p = &sXW[arow * XWK + kt * 32 + acol];
          fa.u[0] = *(const uint4*)p;
          fa.u[1] = *(const uint4*)(p + 16);
        } else {
          const unsigned short* p = hnext + (size_t)arow * HPAD + (kt * 32 - XWK) + acol;
          fa.u[0] = *(const uint4*)p;
          fa.u[1] = *(const uint4*)(p + 16);
        }
        #pragma unroll
        for (int nt = 0; nt < 2; ++nt) {
          FragA fb;
          const unsigned short* q = watt_g + (size_t)(nt * 16 + (lane & 15)) * KDIM
                                    + kt * 32 + bk0;
          fb.u[0] = *(const uint4*)q;
          fb.u[1] = *(const uint4*)(q + 16);
          aat[nt] = __builtin_amdgcn_wmma_f32_16x16x32_bf16(false, fa.v, false, fb.v,
                                                            (short)0, aat[nt], false, false);
        }
      }
      #pragma unroll
      for (int nt = 0; nt < 2; ++nt) {
        int n = nt * 16 + (lane & 15);
        float bias = (n < 3 * Kmix) ? b_att[n] : 0.f;
        #pragma unroll
        for (int r = 0; r < 8; ++r)
          sAtt[(size_t)(mrow + r) * 32 + n] = aat[nt][r] + bias;
      }
      __syncthreads();

      // Gaussian window: 2 threads per batch row
      {
        const int b = tid >> 1, half = tid & 1;
        for (int i = tid; i < Bn * VOCABn; i += 256) sW[i] = 0.f;
        if (half == 0) {
          #pragma unroll
          for (int k = 0; k < Kmix; ++k)
            sKappa[b * Kmix + k] += __expf(sAtt[b * 32 + 2 * Kmix + k]) * 0.04f;
        }
        __syncthreads();
        float al[Kmix], be[Kmix], ka[Kmix];
        #pragma unroll
        for (int k = 0; k < Kmix; ++k) {
          al[k] = __expf(sAtt[b * 32 + k]);
          be[k] = __expf(sAtt[b * 32 + Kmix + k]);
          ka[k] = sKappa[b * Kmix + k];
        }
        const int cl = clen[b];
        for (int u = half * 32; u < half * 32 + 32; ++u) {
          if (u < cl) {
            float phi = 0.f;
            #pragma unroll
            for (int k = 0; k < Kmix; ++k) {
              float du = ka[k] - (float)u;
              phi += al[k] * __expf(-be[k] * du * du);
            }
            atomicAdd(&sW[b * VOCABn + c[b * Un + u]], phi);
          }
        }
        __syncthreads();
        for (int i = tid; i < Bn * VOCABn; i += 256) wbuf[i] = f2bf(sW[i]);
      }
    }

    grid_barrier(ctr, (unsigned)(NWG * (2 * t + 2)));   // w_t published
  }
}

// ---------------------------------------------------------------------------
// Kernel 2: out = hs @ W_out + b_out.  [102400 x 400] @ [400 x 121].
// HBM-bound (82 MB bf16 of hs ~ 4 us at 23.3 TB/s); WMMA bf16, 8 N-tiles/wave.
// W_out (128x416 bf16, 104 KB) is TDM-staged into LDS once per block so the
// B operand streams from ds_load_b128 instead of 8x-redundant L2 reads.
// ---------------------------------------------------------------------------
__global__ __launch_bounds__(256) void out_proj(
    const unsigned short* __restrict__ hs, const unsigned short* __restrict__ wout,
    const float* __restrict__ b_out, float* __restrict__ out)
{
  __shared__ unsigned short sWo[128 * HPAD];   // 104 KB

  const int lane = threadIdx.x & 31;
  const int wave = threadIdx.x >> 5;
  const int mtile = blockIdx.x * 8 + wave;          // 6400 M-tiles
  const int arow = mtile * 16 + (lane & 15);
  const int acol = (lane >> 4) << 3;
  const int bk0  = (lane >> 4) << 3;

  if (wave == 0) {
    tdm_load_tile_2d(&sWo[0], wout, HPAD, 128, HPAD);
    __builtin_amdgcn_s_wait_tensorcnt(0);
  }
  __syncthreads();

  f32x8 acc[8];
  #pragma unroll
  for (int nt = 0; nt < 8; ++nt)
    #pragma unroll
    for (int r = 0; r < 8; ++r) acc[nt][r] = 0.f;

  const unsigned short* rowp = hs + (size_t)arow * Hn;
  __builtin_prefetch(rowp, 0, 0);

  #pragma unroll
  for (int kt = 0; kt < 13; ++kt) {                  // K = 416 (pad cols have zero B)
    FragA fa;
    const unsigned short* p = rowp + kt * 32 + acol;
    fa.u[0] = *(const uint4*)p;
    fa.u[1] = *(const uint4*)(p + 16);
    #pragma unroll
    for (int nt = 0; nt < 8; ++nt) {
      FragA fb;
      const unsigned short* q = &sWo[(nt * 16 + (lane & 15)) * HPAD + kt * 32 + bk0];
      fb.u[0] = *(const uint4*)q;
      fb.u[1] = *(const uint4*)(q + 16);
      acc[nt] = __builtin_amdgcn_wmma_f32_16x16x32_bf16(false, fa.v, false, fb.v,
                                                        (short)0, acc[nt], false, false);
    }
  }

  const int mrow = mtile * 16 + ((lane >> 4) << 3);
  #pragma unroll
  for (int nt = 0; nt < 8; ++nt) {
    int n = nt * 16 + (lane & 15);
    if (n < OUTn) {
      float bias = b_out[n];
      #pragma unroll
      for (int r = 0; r < 8; ++r)
        out[(size_t)(mrow + r) * OUTn + n] = acc[nt][r] + bias;
    }
  }
}

// ---------------------------------------------------------------------------
extern "C" void kernel_launch(void* const* d_in, const int* in_sizes, int n_in,
                              void* d_out, int out_size, void* d_ws, size_t ws_size,
                              hipStream_t stream) {
  (void)in_sizes; (void)n_in; (void)out_size; (void)ws_size;
  const float* x    = (const float*)d_in[0];
  const int*   c    = (const int*)d_in[1];
  const int*   clen = (const int*)d_in[3];
  const float* Wx   = (const float*)d_in[4];
  const float* Wh   = (const float*)d_in[5];
  const float* bl   = (const float*)d_in[6];
  const float* Wa   = (const float*)d_in[7];
  const float* ba   = (const float*)d_in[8];
  const float* Wo   = (const float*)d_in[9];
  const float* bo   = (const float*)d_in[10];
  float* out = (float*)d_out;

  char* ws = (char*)d_ws;
  // State region (memset to zero every launch -> deterministic, capture-safe):
  unsigned*       ctr  = (unsigned*)ws;                                   // 256 B
  unsigned short* hbuf = (unsigned short*)(ws + 256);                     // 2x128x416 bf16
  unsigned short* wbuf = (unsigned short*)(ws + 256 + 2 * Bn * HPAD * 2); // 128x80 bf16
  size_t o = 256 + (size_t)2 * Bn * HPAD * 2 + (size_t)Bn * VOCABn * 2;   // = 233728
  const size_t state_bytes = o;
  unsigned short* wz   = (unsigned short*)(ws + o); o += (size_t)1600 * KDIM * 2;
  unsigned short* watt = (unsigned short*)(ws + o); o += (size_t)32 * KDIM * 2;
  unsigned short* wout = (unsigned short*)(ws + o); o += (size_t)128 * HPAD * 2;
  unsigned short* hs   = (unsigned short*)(ws + o);  // 128*800*400 bf16 (~82 MB)

  (void)hipMemsetAsync(d_ws, 0, state_bytes, stream);
  prep_weights<<<dim3(1024), dim3(256), 0, stream>>>(Wx, Wh, Wa, Wo, wz, watt, wout);
  rnn_persistent<<<dim3(NWG), dim3(256), 0, stream>>>(x, c, clen, bl, ba, wz, watt,
                                                      hbuf, wbuf, hs, ctr);
  out_proj<<<dim3(800), dim3(256), 0, stream>>>(hs, wout, bo, out);
}
